// StructureLearning_54683523613020
// MI455X (gfx1250) — matrix operand order repaired and checked by reading, not compiled, not run on hardware
//
#include <hip/hip_runtime.h>
#include <hip/hip_bf16.h>

typedef __attribute__((ext_vector_type(16))) _Float16 v16h;
typedef __attribute__((ext_vector_type(8)))  _Float16 v8h;
typedef __attribute__((ext_vector_type(8)))  float    v8f;

#define FEA 128
#define WFT 64
#define THRESH 0.5f
#define EPS_N 1e-8f

// ---------------------------------------------------------------------------
// Kernel 1: embed = fea @ w  (row of 64), row-normalize (cosine prep), cast
// to f16. One 64-thread block (2 waves) per embedding row. w (128x64 = 32KB)
// stays hot in L2/WGP$; fea row staged through LDS.
// ---------------------------------------------------------------------------
__global__ __launch_bounds__(64) void sl_embed_norm_f16(
    const float* __restrict__ fea, const float* __restrict__ w,
    _Float16* __restrict__ nout)
{
  __shared__ float s_fea[FEA];
  __shared__ float s_e[WFT];
  const int row = blockIdx.x;
  const int t = threadIdx.x;

  s_fea[t]      = fea[(size_t)row * FEA + t];
  s_fea[t + 64] = fea[(size_t)row * FEA + t + 64];
  __syncthreads();

  float acc = 0.f;
  #pragma unroll 8
  for (int k = 0; k < FEA; ++k)
    acc = fmaf(s_fea[k], w[k * WFT + t], acc);

  s_e[t] = acc;
  __syncthreads();

  float ssq = 0.f;
  #pragma unroll
  for (int j = 0; j < WFT; ++j)
    ssq = fmaf(s_e[j], s_e[j], ssq);

  float nrm = fmaxf(sqrtf(ssq), EPS_N);   // matches max(||a||, EPS)
  nout[(size_t)row * WFT + t] = (_Float16)(acc / nrm);
}

// ---------------------------------------------------------------------------
// Kernel 2: C = threshold( nA (MxK=*x64) @ nB^T ), K = 64.
// Wave32 WMMA: each wave computes a 16x64 strip (4 tiles of 16x16), two
// v_wmma_f32_16x16x32_f16 per tile (K=64), A fragments loaded once per strip.
//
// Fragment layouts used here (16-bit, wave32; same contiguous k order in A and B):
//   A 16x32 : lane<16 -> M=lane,  K = k0+0..15
//             lane>=16-> M=lane-16, K = k0+16..31
//   B 32x16 : lane<16 -> N=lane,  K = k0+0..15 ; lane>=16 -> K = k0+16..31
//   C/D f32 : VGPR r  -> M = r + 8*(lane>=16), N = lane&15
// ---------------------------------------------------------------------------
__global__ __launch_bounds__(256) void sl_sim_wmma_f16(
    const _Float16* __restrict__ nA, const _Float16* __restrict__ nB,
    float* __restrict__ out, int n)
{
  const int wave = threadIdx.x >> 5;
  const int lane = threadIdx.x & 31;
  const int hi   = lane >> 4;      // half-wave select
  const int m    = lane & 15;

  const int stripsPerRow = n >> 6;                 // n/64 strips along N
  const long strip = (long)blockIdx.x * 8 + wave;  // 8 waves per block
  const int it = (int)(strip / stripsPerRow);      // 16-row tile index
  const int js = (int)(strip % stripsPerRow);      // 64-col strip index
  const int i0 = it << 4;
  const int j0 = js << 6;

  // ---- A fragments for K=0..31 (a0) and K=32..63 (a1): 2x16B per frag ----
  const _Float16* arow = nA + (size_t)(i0 + m) * WFT;
  v8h a00 = *(const v8h*)(arow +      hi * 16);
  v8h a01 = *(const v8h*)(arow + 8  + hi * 16);
  v8h a10 = *(const v8h*)(arow + 32 + hi * 16);
  v8h a11 = *(const v8h*)(arow + 40 + hi * 16);
  v16h a0 = __builtin_shufflevector(a00, a01, 0,1,2,3,4,5,6,7,8,9,10,11,12,13,14,15);
  v16h a1 = __builtin_shufflevector(a10, a11, 0,1,2,3,4,5,6,7,8,9,10,11,12,13,14,15);

  #pragma unroll
  for (int t = 0; t < 4; ++t) {
    // B fragment: lane holds row (j0+t*16+m) of nB, 16 contiguous halves.
    const _Float16* brow = nB + (size_t)(j0 + t * 16 + m) * WFT;
    v16h b0 = *(const v16h*)(brow +      hi * 16);   // K = 0..31
    v16h b1 = *(const v16h*)(brow + 32 + hi * 16);   // K = 32..63

    v8f c = {};
    c = __builtin_amdgcn_wmma_f32_16x16x32_f16(false, a0, false, b0,
                                               (short)0, c, false, false);
    c = __builtin_amdgcn_wmma_f32_16x16x32_f16(false, a1, false, b1,
                                               (short)0, c, false, false);

    // Threshold + store. Each half-wave writes 64B contiguous per VGPR row.
    float* ocol = out + (size_t)(j0 + t * 16 + m);
    #pragma unroll
    for (int r = 0; r < 8; ++r) {
      float v = c[r];
      v = (v < THRESH) ? 0.0f : v;
      ocol[(size_t)(i0 + r + 8 * hi) * (size_t)n] = v;
    }
  }
}

extern "C" void kernel_launch(void* const* d_in, const int* in_sizes, int n_in,
                              void* d_out, int out_size, void* d_ws, size_t ws_size,
                              hipStream_t stream) {
  const float* fea1 = (const float*)d_in[0];
  const float* fea2 = (const float*)d_in[1];
  const float* w1   = (const float*)d_in[2];
  const float* w2   = (const float*)d_in[3];

  const int n1 = in_sizes[0] / FEA;   // 8192
  const int n2 = in_sizes[1] / FEA;   // 8192

  _Float16* nrm1 = (_Float16*)d_ws;                       // [n1, 64] f16
  _Float16* nrm2 = nrm1 + (size_t)n1 * WFT;               // [n2, 64] f16

  // Embed + normalize + cast (tiny).
  sl_embed_norm_f16<<<n1, 64, 0, stream>>>(fea1, w1, nrm1);
  sl_embed_norm_f16<<<n2, 64, 0, stream>>>(fea2, w2, nrm2);

  float* out_up = (float*)d_out;                          // [n1, n2]
  float* out_uu = out_up + (size_t)n1 * n2;               // [n1, n1]

  // (n1/16) tile rows * (n2/64) strips, 8 waves (strips) per 256-thread block.
  const int blocks_up = (n1 / 16) * (n2 / 64) / 8;
  sl_sim_wmma_f16<<<blocks_up, 256, 0, stream>>>(nrm1, nrm2, out_up, n2);

  const int blocks_uu = (n1 / 16) * (n1 / 64) / 8;
  sl_sim_wmma_f16<<<blocks_uu, 256, 0, stream>>>(nrm1, nrm1, out_uu, n1);
}
